// VectorQuantizer_13967233646719
// MI455X (gfx1250) — matrix operand order repaired and checked by reading, not compiled, not run on hardware
//
#include <hip/hip_runtime.h>
#include <math.h>

// ---- problem constants (match reference) ----
#define N_ROWS     131072
#define EMBED_DIM  256
#define NUM_CODES  1024
#define COMMIT_COST 0.25f

// d_out layout (flat, return order): zq_st | residual | indices | loss | perplexity
#define ZQ_OFF   ((size_t)0)
#define RES_OFF  ((size_t)N_ROWS * EMBED_DIM)                 // 33,554,432
#define IDX_OFF  ((size_t)2 * N_ROWS * EMBED_DIM)             // 67,108,864
#define LOSS_OFF (IDX_OFF + N_ROWS)                           // 67,239,936
#define PERP_OFF (LOSS_OFF + 1)

typedef __bf16 v16bf __attribute__((ext_vector_type(16)));
typedef __bf16 v8bf  __attribute__((ext_vector_type(8)));
typedef float  v8f   __attribute__((ext_vector_type(8)));
typedef int    v4i   __attribute__((ext_vector_type(4)));

#define TILE_CODES  16
#define CODE_STRIDE 520   // 256 h + 256 l + 8 pad bf16 -> 260 dwords = 4 mod 64: conflict-free b128 reads

// ---- async global->LDS path (CDNA5), with portable fallback --------------
#if defined(__HIP_DEVICE_COMPILE__) && defined(__has_builtin)
#if __has_builtin(__builtin_amdgcn_global_load_async_to_lds_b128) && \
    __has_builtin(__builtin_amdgcn_s_wait_asynccnt)
#define HAVE_ASYNC_LDS 1
#endif
#endif

#ifdef HAVE_ASYNC_LDS
typedef __attribute__((address_space(1))) v4i* as1_v4i_p;
typedef __attribute__((address_space(3))) v4i* as3_v4i_p;
#endif

__device__ __forceinline__ void stage16(const __bf16* g, __bf16* l) {
#ifdef HAVE_ASYNC_LDS
  __builtin_amdgcn_global_load_async_to_lds_b128(
      (as1_v4i_p)(void*)g,   // global source, 16 bytes
      (as3_v4i_p)(void*)l,   // LDS destination
      /*offset=*/0, /*cpol=*/0);
#else
  *(uint4*)l = *(const uint4*)g;   // reg round-trip fallback (ds_store_b128)
#endif
}

__device__ __forceinline__ void stage_wait() {
#ifdef HAVE_ASYNC_LDS
  __builtin_amdgcn_s_wait_asynccnt(0);
#endif
}

// ---------------------------------------------------------------------------
// Kernel 1: c2[k] (exact f32), split codebook into bf16 hi+lo, zero histogram.
// ---------------------------------------------------------------------------
__global__ __launch_bounds__(256)
void vq_prep_kernel(const float* __restrict__ cb,
                    float* __restrict__ c2,
                    unsigned int* __restrict__ counts,
                    __bf16* __restrict__ cbh,
                    __bf16* __restrict__ cbl) {
  int k = blockIdx.x * 256 + threadIdx.x;
  if (k >= NUM_CODES) return;
  const float* row = cb + (size_t)k * EMBED_DIM;
  __bf16* hrow = cbh + (size_t)k * EMBED_DIM;
  __bf16* lrow = cbl + (size_t)k * EMBED_DIM;
  float s = 0.0f;
#pragma unroll 4
  for (int d = 0; d < EMBED_DIM; d += 4) {
    float4 v = *(const float4*)(row + d);
    s += v.x * v.x + v.y * v.y + v.z * v.z + v.w * v.w;
    __bf16 hx = (__bf16)v.x, hy = (__bf16)v.y, hz = (__bf16)v.z, hw = (__bf16)v.w;
    hrow[d] = hx; hrow[d + 1] = hy; hrow[d + 2] = hz; hrow[d + 3] = hw;
    lrow[d]     = (__bf16)(v.x - (float)hx);
    lrow[d + 1] = (__bf16)(v.y - (float)hy);
    lrow[d + 2] = (__bf16)(v.z - (float)hz);
    lrow[d + 3] = (__bf16)(v.w - (float)hw);
  }
  c2[k] = s;
  counts[k] = 0u;
}

// ---------------------------------------------------------------------------
// Kernel 2: main. One wave owns 16 rows; split-bf16 A tile (zh,zl) lives in
// 128 VGPRs. Codebook tiles (16 codes, h+l) are double-buffer staged in LDS
// (async when available) and shared by all 8 waves. Per tile & 32-wide K
// chunk: 3x v_wmma_f32_16x16x32_bf16 (hh + lh + hl ~= f32-accurate dot).
// score = c2[n] - 2*dot, per-row argmin, 16-lane butterfly reduce, gather,
// write zq/residual/indices, deterministic block reduction of sum(res^2).
// ---------------------------------------------------------------------------
__global__ __launch_bounds__(256)
void vq_main_kernel(const float* __restrict__ z,
                    const float* __restrict__ cb,
                    const float* __restrict__ c2g,
                    const __bf16* __restrict__ cbh,
                    const __bf16* __restrict__ cbl,
                    float* __restrict__ partial_sq,
                    unsigned int* __restrict__ counts,
                    float* __restrict__ out) {
  __shared__ __align__(16) __bf16 stage[2][TILE_CODES * CODE_STRIDE];
  __shared__ float red[256];

  const int lane   = threadIdx.x & 31;
  const int wave   = threadIdx.x >> 5;
  const int lane16 = lane & 15;
  const int hi     = lane >> 4;
  const int m0     = blockIdx.x * 128 + wave * 16;

  // ---- build split-bf16 A fragments on the fly from f32 z (held in regs)
  // 16-bit A 16x32 layout: lanes 0-15 els 0-7 -> K0..7, els 8-15 -> K16..23;
  // lanes 16-31 shifted by +8 (hi*8).
  v16bf ah[8], al[8];
  {
    const float* zr = z + (size_t)(m0 + lane16) * EMBED_DIM;
#pragma unroll
    for (int kk = 0; kk < 8; ++kk) {
      const int ko = kk * 32 + hi * 8;
      float4 f0 = *(const float4*)(zr + ko);
      float4 f1 = *(const float4*)(zr + ko + 4);
      float4 g0 = *(const float4*)(zr + ko + 16);
      float4 g1 = *(const float4*)(zr + ko + 20);
      float fv[8] = {f0.x, f0.y, f0.z, f0.w, f1.x, f1.y, f1.z, f1.w};
      float gv[8] = {g0.x, g0.y, g0.z, g0.w, g1.x, g1.y, g1.z, g1.w};
#pragma unroll
      for (int e = 0; e < 8; ++e) {
        __bf16 h = (__bf16)fv[e];
        ah[kk][e] = h;
        al[kk][e] = (__bf16)(fv[e] - (float)h);
        __bf16 h2 = (__bf16)gv[e];
        ah[kk][8 + e] = h2;
        al[kk][8 + e] = (__bf16)(gv[e] - (float)h2);
      }
    }
  }

  float best_s[8];
  int   best_i[8];
#pragma unroll
  for (int r = 0; r < 8; ++r) { best_s[r] = 3.4e38f; best_i[r] = 0; }

  // cooperative tile stager: 16 threads per code, 32B (16 elems) per thread
  auto stage_tile = [&](int t, int buf) {
    const int code = threadIdx.x >> 4;
    const int sub  = threadIdx.x & 15;
    const size_t gro = (size_t)(t * TILE_CODES + code) * EMBED_DIM + sub * 16;
    __bf16* lh = &stage[buf][code * CODE_STRIDE + sub * 16];
    __bf16* ll = &stage[buf][code * CODE_STRIDE + 256 + sub * 16];
    stage16(cbh + gro,     lh);
    stage16(cbh + gro + 8, lh + 8);
    stage16(cbl + gro,     ll);
    stage16(cbl + gro + 8, ll + 8);
  };

  stage_tile(0, 0);
  stage_wait();
  __syncthreads();

  // ---- sweep all 1024 codes in 16-wide tiles (double-buffered LDS)
  for (int t = 0; t < NUM_CODES / TILE_CODES; ++t) {
    const int cur = t & 1;
    if (t + 1 < NUM_CODES / TILE_CODES) stage_tile(t + 1, cur ^ 1);

    const int n = t * TILE_CODES + lane16;       // this lane's code column
    const float c2n = c2g[n];
    // dense-B 32x16 layout: lanes 0-15 els 0-7 -> K0..7, els 8-15 -> K16..23; lanes 16-31 +8
    const __bf16* bbase = &stage[cur][lane16 * CODE_STRIDE];

    v8f acc = {};
#pragma unroll
    for (int kk = 0; kk < 8; ++kk) {
      const int ko = kk * 32 + hi * 8;
      v8bf bh0 = *(const v8bf*)(bbase + ko);
      v8bf bh1 = *(const v8bf*)(bbase + ko + 16);
      v8bf bl0 = *(const v8bf*)(bbase + 256 + ko);
      v8bf bl1 = *(const v8bf*)(bbase + 256 + ko + 16);
      v16bf bh = __builtin_shufflevector(bh0, bh1, 0,1,2,3,4,5,6,7,8,9,10,11,12,13,14,15);
      v16bf bl = __builtin_shufflevector(bl0, bl1, 0,1,2,3,4,5,6,7,8,9,10,11,12,13,14,15);
      acc = __builtin_amdgcn_wmma_f32_16x16x32_bf16(false, ah[kk], false, bh, (short)0, acc, false, false);
      acc = __builtin_amdgcn_wmma_f32_16x16x32_bf16(false, al[kk], false, bh, (short)0, acc, false, false);
      acc = __builtin_amdgcn_wmma_f32_16x16x32_bf16(false, ah[kk], false, bl, (short)0, acc, false, false);
    }

#pragma unroll
    for (int r = 0; r < 8; ++r) {
      float sc = c2n - 2.0f * acc[r];            // row m0 + r + 8*hi, column n
      if (sc < best_s[r] || (sc == best_s[r] && n < best_i[r])) {
        best_s[r] = sc; best_i[r] = n;
      }
    }

    stage_wait();
    __syncthreads();
  }

  // ---- argmin across the 16 lanes of each half-wave (rows 0-7 | 8-15)
#pragma unroll
  for (int r = 0; r < 8; ++r) {
#pragma unroll
    for (int d = 8; d >= 1; d >>= 1) {
      float os = __shfl_xor(best_s[r], d, 16);
      int   oi = __shfl_xor(best_i[r], d, 16);
      if (os < best_s[r] || (os == best_s[r] && oi < best_i[r])) {
        best_s[r] = os; best_i[r] = oi;
      }
    }
  }

  // ---- gather + write outputs for the wave's 16 rows (exact f32)
  float sq = 0.0f;
#pragma unroll 1
  for (int j = 0; j < 16; ++j) {
    const int idx = __shfl(best_i[j & 7], (j >> 3) << 4, 32);
    const int m = m0 + j;
    if (lane == 0) {
      out[IDX_OFF + (size_t)m] = (float)idx;
      atomicAdd(counts + idx, 1u);
    }
    const float4* zr = (const float4*)(z  + (size_t)m   * EMBED_DIM) + lane * 2;
    const float4* cr = (const float4*)(cb + (size_t)idx * EMBED_DIM) + lane * 2;
    float4* q  = (float4*)(out + ZQ_OFF  + (size_t)m * EMBED_DIM) + lane * 2;
    float4* rs = (float4*)(out + RES_OFF + (size_t)m * EMBED_DIM) + lane * 2;
#pragma unroll
    for (int h = 0; h < 2; ++h) {
      float4 zv = zr[h];
      float4 cv = cr[h];
      float4 rv = make_float4(zv.x - cv.x, zv.y - cv.y, zv.z - cv.z, zv.w - cv.w);
      q[h]  = cv;
      rs[h] = rv;
      sq += rv.x * rv.x + rv.y * rv.y + rv.z * rv.z + rv.w * rv.w;
    }
  }

  // ---- deterministic per-block reduction of sum(residual^2)
  red[threadIdx.x] = sq;
  __syncthreads();
#pragma unroll
  for (int s2 = 128; s2 > 0; s2 >>= 1) {
    if ((int)threadIdx.x < s2) red[threadIdx.x] += red[threadIdx.x + s2];
    __syncthreads();
  }
  if (threadIdx.x == 0) partial_sq[blockIdx.x] = red[0];
}

// ---------------------------------------------------------------------------
// Kernel 3: fixed-order finalize -> commitment loss + perplexity scalars.
// ---------------------------------------------------------------------------
__global__ __launch_bounds__(256)
void vq_finalize_kernel(const float* __restrict__ partial_sq,
                        const unsigned int* __restrict__ counts,
                        float* __restrict__ out) {
  __shared__ float red[256];

  float s = 0.0f;
  for (int i = threadIdx.x; i < 1024; i += 256) s += partial_sq[i];
  red[threadIdx.x] = s;
  __syncthreads();
#pragma unroll
  for (int k = 128; k > 0; k >>= 1) {
    if ((int)threadIdx.x < k) red[threadIdx.x] += red[threadIdx.x + k];
    __syncthreads();
  }
  const float total_sq = red[0];
  __syncthreads();

  float ent = 0.0f;
  for (int i = threadIdx.x; i < NUM_CODES; i += 256) {
    float p = (float)counts[i] / (float)N_ROWS;
    ent += p * logf(p + 1e-10f);
  }
  red[threadIdx.x] = ent;
  __syncthreads();
#pragma unroll
  for (int k = 128; k > 0; k >>= 1) {
    if ((int)threadIdx.x < k) red[threadIdx.x] += red[threadIdx.x + k];
    __syncthreads();
  }
  if (threadIdx.x == 0) {
    out[LOSS_OFF] = COMMIT_COST * total_sq / ((float)N_ROWS * (float)EMBED_DIM);
    out[PERP_OFF] = expf(-red[0]);
  }
}

// ---------------------------------------------------------------------------
extern "C" void kernel_launch(void* const* d_in, const int* in_sizes, int n_in,
                              void* d_out, int out_size, void* d_ws, size_t ws_size,
                              hipStream_t stream) {
  const float* z  = (const float*)d_in[0];   // [131072, 256] f32
  const float* cb = (const float*)d_in[1];   // [1024, 256] f32
  float* out = (float*)d_out;

  // workspace: c2[1024] f32 | partial_sq[1024] f32 | counts[1024] u32 |
  //            cbh[1024*256] bf16 | cbl[1024*256] bf16   (~1.04 MB total)
  float* c2            = (float*)d_ws;
  float* partial_sq    = c2 + NUM_CODES;
  unsigned int* counts = (unsigned int*)(partial_sq + 1024);
  __bf16* cbh          = (__bf16*)(counts + 1024);
  __bf16* cbl          = cbh + (size_t)NUM_CODES * EMBED_DIM;

  vq_prep_kernel<<<NUM_CODES / 256, 256, 0, stream>>>(cb, c2, counts, cbh, cbl);
  vq_main_kernel<<<N_ROWS / 128, 256, 0, stream>>>(z, cb, c2, cbh, cbl,
                                                   partial_sq, counts, out);
  vq_finalize_kernel<<<1, 256, 0, stream>>>(partial_sq, counts, out);
}
